// KashfModel_21973052686460
// MI455X (gfx1250) — compile-verified, matrix operand order checked by reading, not run on hardware
//
#include <hip/hip_runtime.h>
#include <hip/hip_bf16.h>
#include <math.h>

typedef __attribute__((ext_vector_type(16))) __bf16 v16bf;
typedef __attribute__((ext_vector_type(8)))  __bf16 v8bf;
typedef __attribute__((ext_vector_type(8)))  float  v8f;
typedef __attribute__((ext_vector_type(4)))  float  v4f;

#define NTOK 2048      // B*T
#define DMODEL 1024
#define EDIM 512
#define SEQT 1024
#define NHEAD 16
#define HDIM 64
#define EXPD 1024
#define DFF 1365
#define LD1 1408       // padded row stride for DFF buffers (mult of 4)
#define NVOCAB 50257

// ---- WMMA fragment K-index mapping (ISA 7.12.2, 16-bit A 16x32) ----
// e=0..7  -> k = half*8 + e          (contiguous 8)
// e=8..15 -> k = 16 + half*8 + (e-8) (contiguous 8)
__device__ __forceinline__ int klocal(int e, int half) {
    int base = (e < 8) ? e : (8 + e);     // 0..7 -> 0..7 ; 8..15 -> 16..23
    return base + half * 8;
}

// fast fragment load from a contiguous fp32 row (16B-aligned, no guards)
__device__ __forceinline__ v16bf frag_fast(const float* base, int k0, int half) {
    const v4f* p0 = (const v4f*)(base + k0 + half * 8);
    const v4f* p1 = (const v4f*)(base + k0 + 16 + half * 8);
    v4f f0 = p0[0], f1 = p0[1], f2 = p1[0], f3 = p1[1];
    v16bf a;
#pragma unroll
    for (int j = 0; j < 4; ++j) {
        a[j]      = (__bf16)f0[j];
        a[4 + j]  = (__bf16)f1[j];
        a[8 + j]  = (__bf16)f2[j];
        a[12 + j] = (__bf16)f3[j];
    }
    return a;
}

// guarded fragment load (row validity + K tail)
__device__ __forceinline__ v16bf frag_guarded(const float* base, int k0, int half,
                                              int K, bool av) {
    if (av && (k0 + 31 < K)) return frag_fast(base, k0, half);
    v16bf a;
#pragma unroll
    for (int e = 0; e < 16; ++e) {
        int kk = k0 + klocal(e, half);
        float v = (av && kk < K) ? base[kk] : 0.f;
        a[e] = (__bf16)v;
    }
    return a;
}

__device__ __forceinline__ v16bf cat8(v8bf lo, v8bf hi) {
    return __builtin_shufflevector(lo, hi, 0, 1, 2, 3, 4, 5, 6, 7,
                                   8, 9, 10, 11, 12, 13, 14, 15);
}

union bfpack { __bf16 h[2]; unsigned u; };

// =====================================================================
// Generic GEMM:  C[M x Nc] (+)= A[M x K] * B[K x Nc], fp32 in/out,
// bf16 WMMA compute.  Optional row gather/scatter (MoE routing).
// block = 128 threads (4 waves); wave -> 32x64 tile (8 WMMA / K-step);
// B tile staged column-major in LDS (fragments = 2x ds_load_b128),
// double-buffered so staging of step n+1 overlaps WMMA of step n.
// =====================================================================
__global__ __launch_bounds__(128)
void gemm_bf16_wmma(const float* __restrict__ A, int lda,
                    const float* __restrict__ Bm, int ldb,
                    float* __restrict__ C, int ldc,
                    int M, int Nc, int K, int addTo,
                    const int* __restrict__ rowIdx,
                    const int* __restrict__ rowCnt)
{
    __shared__ __bf16 Bst[2][64][40];  // [buf][local col][k] ; 80B row stride
    int tid  = threadIdx.x;
    int wave = tid >> 5, lane = tid & 31;
    int half = lane >> 4, l15 = lane & 15;

    int mEff = rowIdx ? rowCnt[0] : M;
    if ((int)blockIdx.y * 128 >= mEff) return;   // block-uniform early out

    int m0 = blockIdx.y * 128 + wave * 32;
    int n0 = blockIdx.x * 64;

    int  ar0 = m0 + l15,        ar1 = m0 + 16 + l15;
    bool av0 = ar0 < mEff,      av1 = ar1 < mEff;
    long aRow0 = av0 ? (rowIdx ? (long)rowIdx[ar0] : (long)ar0) : 0;
    long aRow1 = av1 ? (rowIdx ? (long)rowIdx[ar1] : (long)ar1) : 0;
    const float* ab0 = A + aRow0 * lda;
    const float* ab1 = A + aRow1 * lda;

    v8f acc[2][4];
#pragma unroll
    for (int s = 0; s < 2; ++s)
#pragma unroll
        for (int nt = 0; nt < 4; ++nt)
#pragma unroll
            for (int j = 0; j < 8; ++j) acc[s][nt][j] = 0.f;

    int kp = tid >> 3;          // 0..15 : pair of k rows (2*kp, 2*kp+1)
    int c0 = (tid & 7) * 8;     // 0,8,...,56 : 8 local cols

    // stage one 32x64 B tile (column-major, packed b32 writes)
    auto stage = [&](int buf, int k0) {
        int kk0 = k0 + 2 * kp;
        const float* r0 = Bm + (long)kk0 * ldb;
        const float* r1 = r0 + ldb;
        if ((k0 + 32 <= K) && (n0 + 64 <= Nc)) {     // uniform interior path
            float va[8], vb[8];
#pragma unroll
            for (int c = 0; c < 8; ++c) va[c] = r0[n0 + c0 + c];
#pragma unroll
            for (int c = 0; c < 8; ++c) vb[c] = r1[n0 + c0 + c];
#pragma unroll
            for (int c = 0; c < 8; ++c) {
                bfpack pk;
                pk.h[0] = (__bf16)va[c];
                pk.h[1] = (__bf16)vb[c];
                *(unsigned*)&Bst[buf][c0 + c][2 * kp] = pk.u;
            }
            __builtin_prefetch(&r0[(long)32 * ldb + n0 + c0], 0, 1);
        } else {                                      // tail blocks only
            bool v0ok = kk0 < K, v1ok = (kk0 + 1) < K;
#pragma unroll
            for (int c = 0; c < 8; ++c) {
                int nn = n0 + c0 + c;
                bool nok = nn < Nc;
                bfpack pk;
                pk.h[0] = (__bf16)((v0ok && nok) ? r0[nn] : 0.f);
                pk.h[1] = (__bf16)((v1ok && nok) ? r1[nn] : 0.f);
                *(unsigned*)&Bst[buf][c0 + c][2 * kp] = pk.u;
            }
        }
    };

    stage(0, 0);
    int cur = 0;
    for (int k0 = 0; k0 < K; k0 += 32) {
        __syncthreads();
        if (k0 + 32 < K) stage(cur ^ 1, k0 + 32);   // overlap next-step staging

        v16bf a0 = frag_guarded(ab0, k0, half, K, av0);
        v16bf a1 = frag_guarded(ab1, k0, half, K, av1);
#pragma unroll
        for (int nt = 0; nt < 4; ++nt) {
            int col = nt * 16 + l15;
            v8bf lo = *(const v8bf*)&Bst[cur][col][half * 8];
            v8bf hi = *(const v8bf*)&Bst[cur][col][16 + half * 8];
            v16bf b = cat8(lo, hi);
            acc[0][nt] = __builtin_amdgcn_wmma_f32_16x16x32_bf16(
                false, a0, false, b, (short)0, acc[0][nt], false, false);
            acc[1][nt] = __builtin_amdgcn_wmma_f32_16x16x32_bf16(
                false, a1, false, b, (short)0, acc[1][nt], false, false);
        }
        cur ^= 1;
    }

#pragma unroll
    for (int s = 0; s < 2; ++s) {
#pragma unroll
        for (int r = 0; r < 8; ++r) {
            int gm = m0 + s * 16 + r + half * 8;
            if (gm < mEff) {
                long crow = rowIdx ? (long)rowIdx[gm] : (long)gm;
#pragma unroll
                for (int nt = 0; nt < 4; ++nt) {
                    int nn = n0 + nt * 16 + l15;
                    if (nn < Nc) {
                        long idx = crow * (long)ldc + nn;
                        float v = acc[s][nt][r];
                        if (addTo) v += C[idx];
                        C[idx] = v;
                    }
                }
            }
        }
    }
}

// =====================================================================
// Flash-style fused attention. One wave per (b, h, 16-query tile).
// Online softmax with tanh soft-cap; QK^T and P*V via bf16 WMMA.
// P re-laid out (C-layout -> A-layout) through LDS; reads are b128.
// Causality + T%32==0 guarantee all key/value indices are in range.
// =====================================================================
__global__ __launch_bounds__(32)
void attn_fused(const float* __restrict__ Q, const float* __restrict__ Kb,
                const float* __restrict__ Vb, float* __restrict__ O)
{
    int q0 = blockIdx.x * 16;
    int hh = blockIdx.y;
    int bb = blockIdx.z;
    int lane = threadIdx.x;
    int half = lane >> 4, l15 = lane & 15;
    __shared__ __bf16 Ps[16][40];    // [query row][k] ; 80B stride

    const float* qp = Q + (long)(bb * SEQT + q0 + l15) * DMODEL + hh * HDIM;
    v16bf aq0 = frag_fast(qp, 0, half);
    v16bf aq1 = frag_fast(qp, 32, half);

    v8f zero;
#pragma unroll
    for (int j = 0; j < 8; ++j) zero[j] = 0.f;
    v8f o[4] = { zero, zero, zero, zero };
    float rmax[8], rsum[8];
#pragma unroll
    for (int r = 0; r < 8; ++r) { rmax[r] = -1e30f; rsum[r] = 0.f; }

    for (int j0 = 0; j0 < q0 + 16; j0 += 32) {
        float p0[8], p1[8];
#pragma unroll
        for (int s = 0; s < 2; ++s) {
            const float* kp = Kb + (long)(bb * SEQT + j0 + s * 16 + l15) * HDIM;
            v16bf bk0 = frag_fast(kp, 0, half);
            v16bf bk1 = frag_fast(kp, 32, half);
            v8f sa = zero;
            sa = __builtin_amdgcn_wmma_f32_16x16x32_bf16(false, aq0, false, bk0, (short)0, sa, false, false);
            sa = __builtin_amdgcn_wmma_f32_16x16x32_bf16(false, aq1, false, bk1, (short)0, sa, false, false);
            float* pp = s ? p1 : p0;
#pragma unroll
            for (int r = 0; r < 8; ++r) {
                int qm = q0 + r + half * 8;
                int j  = j0 + s * 16 + l15;
                float v = sa[r] * 0.125f;              // HD^-0.5
                v = tanhf(v * 0.02f) * 50.f;           // soft cap
                if (j > qm) v = -1e30f;                // causal mask
                pp[r] = v;
            }
        }
#pragma unroll
        for (int r = 0; r < 8; ++r) {
            float mx = fmaxf(p0[r], p1[r]);
#pragma unroll
            for (int d = 1; d < 16; d <<= 1) mx = fmaxf(mx, __shfl_xor(mx, d, 32));
            float mnew = fmaxf(rmax[r], mx);
            float f  = __expf(rmax[r] - mnew);
            float e0 = __expf(p0[r] - mnew);
            float e1 = __expf(p1[r] - mnew);
            p0[r] = e0; p1[r] = e1;
            float sm = e0 + e1;
#pragma unroll
            for (int d = 1; d < 16; d <<= 1) sm += __shfl_xor(sm, d, 32);
            rsum[r] = rsum[r] * f + sm;
            rmax[r] = mnew;
#pragma unroll
            for (int nt = 0; nt < 4; ++nt) o[nt][r] *= f;
        }
        __syncthreads();
#pragma unroll
        for (int r = 0; r < 8; ++r) {
            Ps[r + half * 8][l15]      = (__bf16)p0[r];
            Ps[r + half * 8][16 + l15] = (__bf16)p1[r];
        }
        __syncthreads();
        v8bf plo = *(const v8bf*)&Ps[l15][half * 8];
        v8bf phi = *(const v8bf*)&Ps[l15][16 + half * 8];
        v16bf ap = cat8(plo, phi);
#pragma unroll
        for (int nt = 0; nt < 4; ++nt) {
            v16bf bv;
#pragma unroll
            for (int e = 0; e < 16; ++e) {
                int tv = j0 + klocal(e, half);
                bv[e] = (__bf16)Vb[(long)(bb * SEQT + tv) * HDIM + nt * 16 + l15];
            }
            o[nt] = __builtin_amdgcn_wmma_f32_16x16x32_bf16(false, ap, false, bv, (short)0, o[nt], false, false);
        }
    }
#pragma unroll
    for (int r = 0; r < 8; ++r) {
        float inv = 1.f / rsum[r];
        long row = (long)(bb * SEQT + q0 + r + half * 8) * DMODEL + hh * HDIM;
#pragma unroll
        for (int nt = 0; nt < 4; ++nt)
            O[row + nt * 16 + l15] = o[nt][r] * inv;
    }
}

// ---------------------------------------------------------------------
// Small kernels
// ---------------------------------------------------------------------
__global__ __launch_bounds__(256)
void rmsnorm_k(const float* __restrict__ X, const float* __restrict__ W,
               float* __restrict__ Y, int Dd)
{
    __shared__ float red[256];
    int n = blockIdx.x;
    const float* x = X + (long)n * Dd;
    float s = 0.f;
    for (int d = threadIdx.x; d < Dd; d += 256) { float v = x[d]; s += v * v; }
    red[threadIdx.x] = s; __syncthreads();
    for (int ofs = 128; ofs > 0; ofs >>= 1) {
        if (threadIdx.x < ofs) red[threadIdx.x] += red[threadIdx.x + ofs];
        __syncthreads();
    }
    float r = rsqrtf(red[0] / (float)Dd + 1e-6f);
    float* y = Y + (long)n * Dd;
    for (int d = threadIdx.x; d < Dd; d += 256) y[d] = x[d] * r * W[d];
}

// fused per-head RMSNorm + RoPE (in place).  32 threads = 32 (even,odd) pairs.
__global__ __launch_bounds__(32)
void rms_rope_k(float* __restrict__ Q, const float* __restrict__ Wn,
                int NH, int rowstride)
{
    int bid = blockIdx.x;
    int n = bid / NH, hh = bid % NH;
    float* p = Q + (long)n * rowstride + hh * HDIM;
    int i = threadIdx.x;
    float a = p[2 * i], b = p[2 * i + 1];
    float s = a * a + b * b;
    for (int d = 1; d < 32; d <<= 1) s += __shfl_xor(s, d, 32);
    float r  = rsqrtf(s / 64.f + 1e-6f);
    float an = a * r * Wn[2 * i];
    float bn = b * r * Wn[2 * i + 1];
    int pos = n % SEQT;
    float inv = __expf(-((float)(2 * i) / 64.f) * 13.122363377404328f); // ln(5e5)
    float ang = (float)pos * inv;
    float c = __cosf(ang), sn = __sinf(ang);
    p[2 * i]     = an * c - bn * sn;
    p[2 * i + 1] = an * sn + bn * c;
}

__global__ __launch_bounds__(256)
void gather_k(const int* __restrict__ tok, const float* __restrict__ Et,
              float* __restrict__ out)
{
    int n = blockIdx.x;
    int t = tok[n];
    for (int d = threadIdx.x; d < EDIM; d += 256)
        out[(long)n * EDIM + d] = Et[(long)t * EDIM + d];
}

__global__ void zero2_k(int* c) { if (threadIdx.x < 2) c[threadIdx.x] = 0; }

__global__ __launch_bounds__(32)
void route_k(const float* __restrict__ X, const float* __restrict__ R,
             const float* __restrict__ rb, int* __restrict__ cnt,
             int* __restrict__ lists)
{
    int n = blockIdx.x;
    const float* x = X + (long)n * DMODEL;
    float a0 = 0.f, a1 = 0.f;
    for (int d = threadIdx.x; d < DMODEL; d += 32) {
        float v = x[d];
        a0 += v * R[d * 2 + 0];
        a1 += v * R[d * 2 + 1];
    }
    for (int d = 1; d < 32; d <<= 1) {
        a0 += __shfl_xor(a0, d, 32);
        a1 += __shfl_xor(a1, d, 32);
    }
    if (threadIdx.x == 0) {
        int e = (a1 + rb[1] > a0 + rb[0]) ? 1 : 0;   // argmax, first-wins tie
        int pos = atomicAdd(&cnt[e], 1);
        lists[e * NTOK + pos] = n;
    }
}

__global__ __launch_bounds__(256)
void silu_mul_k(float* __restrict__ A, const float* __restrict__ Bv,
                long total, int cols, int ld)
{
    for (long i = (long)blockIdx.x * 256 + threadIdx.x; i < total;
         i += (long)gridDim.x * 256) {
        long rr = i / cols, cc = i % cols;
        long idx = rr * ld + cc;
        float g = A[idx], u = Bv[idx];
        A[idx] = g / (1.f + __expf(-g)) * u;
    }
}

__global__ __launch_bounds__(256)
void loopemb_k(float* __restrict__ x, const float* __restrict__ h, float tf)
{
    long i = (long)blockIdx.x * 256 + threadIdx.x;
    if (i >= (long)NTOK * DMODEL) return;
    int d = (int)(i & (DMODEL - 1));
    float emb = 0.f;
    if (d < 256) {
        float f = __expf(-((float)d / 256.f) * 9.2103403719761836f); // ln(1e4)
        emb = __sinf(tf * f);
    } else if (d < 512) {
        float f = __expf(-((float)(d - 256) / 256.f) * 9.2103403719761836f);
        emb = __cosf(tf * f);
    }
    x[i] = h[i] + emb;
}

__global__ __launch_bounds__(256)
void lti_k(float* __restrict__ h, const float* __restrict__ e,
           const float* __restrict__ x, const float* __restrict__ logA,
           const float* __restrict__ logdt, const float* __restrict__ Bv,
           const float* __restrict__ gate, int t)
{
    long i = (long)blockIdx.x * 256 + threadIdx.x;
    if (i >= (long)NTOK * DMODEL) return;
    int d = (int)(i & (DMODEL - 1));
    float z = logdt[0] + logA[d];
    z = fminf(fmaxf(z, -20.f), 20.f);
    float a = __expf(-__expf(z));
    float g = 1.f / (1.f + __expf(-gate[t]));
    h[i] = a * h[i] + Bv[d] * e[i] + g * x[i];
}

// ---------------------------------------------------------------------
// Host-side helpers
// ---------------------------------------------------------------------
static inline void gemm(const float* A, int lda, const float* Bm, int ldb,
                        float* C, int ldc, int M, int Nc, int K, int addTo,
                        const int* rowIdx, const int* rowCnt, hipStream_t s)
{
    dim3 grid((Nc + 63) / 64, (M + 127) / 128);
    gemm_bf16_wmma<<<grid, dim3(128), 0, s>>>(A, lda, Bm, ldb, C, ldc,
                                              M, Nc, K, addTo, rowIdx, rowCnt);
}

static inline void run_attn(const float* xin, float* target,
                            const float* an, const float* wq, const float* wk,
                            const float* wv, const float* wo, const float* qn,
                            const float* kn, float* t0, float* qb, float* kb,
                            float* vb, float* t3, hipStream_t s)
{
    rmsnorm_k<<<NTOK, 256, 0, s>>>(xin, an, t0, DMODEL);
    gemm(t0, DMODEL, wq, DMODEL, qb, DMODEL, NTOK, DMODEL, DMODEL, 0, nullptr, nullptr, s);
    gemm(t0, DMODEL, wk, HDIM,   kb, HDIM,   NTOK, HDIM,   DMODEL, 0, nullptr, nullptr, s);
    gemm(t0, DMODEL, wv, HDIM,   vb, HDIM,   NTOK, HDIM,   DMODEL, 0, nullptr, nullptr, s);
    rms_rope_k<<<NTOK * NHEAD, 32, 0, s>>>(qb, qn, NHEAD, DMODEL);
    rms_rope_k<<<NTOK, 32, 0, s>>>(kb, kn, 1, HDIM);
    attn_fused<<<dim3(SEQT / 16, NHEAD, 2), 32, 0, s>>>(qb, kb, vb, t3);
    gemm(t3, DMODEL, wo, DMODEL, target, DMODEL, NTOK, DMODEL, DMODEL, 1, nullptr, nullptr, s);
}

static inline void run_swiglu(const float* xin, float* target, const float* fn,
                              const float* g, const float* u, const float* dW,
                              float* t0, float* t1, float* t2, hipStream_t s)
{
    rmsnorm_k<<<NTOK, 256, 0, s>>>(xin, fn, t0, DMODEL);
    gemm(t0, DMODEL, g, DFF, t1, LD1, NTOK, DFF, DMODEL, 0, nullptr, nullptr, s);
    gemm(t0, DMODEL, u, DFF, t2, LD1, NTOK, DFF, DMODEL, 0, nullptr, nullptr, s);
    long tot = (long)NTOK * DFF;
    silu_mul_k<<<(int)((tot + 255) / 256), 256, 0, s>>>(t1, t2, tot, DFF, LD1);
    gemm(t1, LD1, dW, DMODEL, target, DMODEL, NTOK, DMODEL, DFF, 1, nullptr, nullptr, s);
}

extern "C" void kernel_launch(void* const* d_in, const int* in_sizes, int n_in,
                              void* d_out, int out_size, void* d_ws, size_t ws_size,
                              hipStream_t stream)
{
    (void)in_sizes; (void)n_in; (void)out_size; (void)ws_size;
#define FIN(i) ((const float*)d_in[i])
    const int*   tok         = (const int*)d_in[0];
    const float* embed_table = FIN(1);
    const float* embed_proj  = FIN(2);
    const float* pre_an = FIN(3),  * pre_fn = FIN(4);
    const float* pre_wq = FIN(5),  * pre_wk = FIN(6),  * pre_wv = FIN(7), * pre_wo = FIN(8);
    const float* pre_qn = FIN(9),  * pre_kn = FIN(10);
    const float* pre_g  = FIN(11), * pre_u  = FIN(12), * pre_d  = FIN(13);
    const float* rec_an = FIN(14), * rec_fn = FIN(15);
    const float* rec_wq = FIN(16), * rec_wk = FIN(17), * rec_wv = FIN(18), * rec_wo = FIN(19);
    const float* rec_qn = FIN(20), * rec_kn = FIN(21);
    const float* rec_router = FIN(22), * rec_rbias = FIN(23);
    const float* rec_eg = FIN(24), * rec_eu = FIN(25), * rec_ed = FIN(26);
    const float* rec_sg = FIN(27), * rec_su = FIN(28), * rec_sd = FIN(29);
    const float* lti_logA = FIN(30), * lti_logdt = FIN(31), * lti_B = FIN(32);
    const float* gate_emb = FIN(33);
    const float* coda_an = FIN(34), * coda_fn = FIN(35);
    const float* coda_wq = FIN(36), * coda_wk = FIN(37), * coda_wv = FIN(38), * coda_wo = FIN(39);
    const float* coda_qn = FIN(40), * coda_kn = FIN(41);
    const float* coda_g  = FIN(42), * coda_u  = FIN(43), * coda_d  = FIN(44);
    const float* final_norm = FIN(45);
    const float* lm_head    = FIN(46);
#undef FIN

    // ---- workspace carve (fp32) ----
    float* f = (float*)d_ws;
    float* e_ = f; f += (long)NTOK * DMODEL;
    float* h_ = f; f += (long)NTOK * DMODEL;
    float* x_ = f; f += (long)NTOK * DMODEL;
    float* t0 = f; f += (long)NTOK * DMODEL;
    float* t3 = f; f += (long)NTOK * DMODEL;
    float* t1 = f; f += (long)NTOK * LD1;
    float* t2 = f; f += (long)NTOK * LD1;
    float* kb = f; f += (long)NTOK * HDIM;
    float* vb = f; f += (long)NTOK * HDIM;
    int* cnt   = (int*)f;
    int* lists = cnt + 2;

    // ---- embedding: e = embed_table[tok] @ embed_proj ----
    gather_k<<<NTOK, 256, 0, stream>>>(tok, embed_table, t1);
    gemm(t1, EDIM, embed_proj, DMODEL, e_, DMODEL, NTOK, DMODEL, EDIM, 0, nullptr, nullptr, stream);
    hipMemcpyAsync(h_, e_, (long)NTOK * DMODEL * sizeof(float),
                   hipMemcpyDeviceToDevice, stream);

    // ---- pre block ----
    run_attn(h_, h_, pre_an, pre_wq, pre_wk, pre_wv, pre_wo, pre_qn, pre_kn,
             t0, t1, kb, vb, t3, stream);
    run_swiglu(h_, h_, pre_fn, pre_g, pre_u, pre_d, t0, t1, t2, stream);

    // ---- recurrent loop ----
    const int ewgrid = (int)(((long)NTOK * DMODEL + 255) / 256);
    for (int t = 0; t < 6; ++t) {
        loopemb_k<<<ewgrid, 256, 0, stream>>>(x_, h_, (float)t);
        run_attn(x_, x_, rec_an, rec_wq, rec_wk, rec_wv, rec_wo, rec_qn, rec_kn,
                 t0, t1, kb, vb, t3, stream);
        rmsnorm_k<<<NTOK, 256, 0, stream>>>(x_, rec_fn, t0, DMODEL);

        // top-1 routing (STE collapses to hard one-hot in forward)
        zero2_k<<<1, 32, 0, stream>>>(cnt);
        route_k<<<NTOK, 32, 0, stream>>>(t0, rec_router, rec_rbias, cnt, lists);

        // shared expert (dense)
        gemm(t0, DMODEL, rec_sg, EXPD, t1, LD1, NTOK, EXPD, DMODEL, 0, nullptr, nullptr, stream);
        gemm(t0, DMODEL, rec_su, EXPD, t2, LD1, NTOK, EXPD, DMODEL, 0, nullptr, nullptr, stream);
        {
            long tot = (long)NTOK * EXPD;
            silu_mul_k<<<(int)((tot + 255) / 256), 256, 0, stream>>>(t1, t2, tot, EXPD, LD1);
        }
        gemm(t1, LD1, rec_sd, DMODEL, x_, DMODEL, NTOK, DMODEL, EXPD, 1, nullptr, nullptr, stream);

        // routed experts (grouped GEMM via row gather/scatter)
        for (int ex = 0; ex < 2; ++ex) {
            const int* ridx = lists + ex * NTOK;
            const int* rc   = cnt + ex;
            const float* egw = rec_eg + (long)ex * DMODEL * EXPD;
            const float* euw = rec_eu + (long)ex * DMODEL * EXPD;
            gemm(t0, DMODEL, egw, EXPD, t1, LD1, NTOK, EXPD, DMODEL, 0, ridx, rc, stream);
            gemm(t0, DMODEL, euw, EXPD, t2, LD1, NTOK, EXPD, DMODEL, 0, ridx, rc, stream);
        }
        {
            long tot = (long)NTOK * EXPD;
            silu_mul_k<<<(int)((tot + 255) / 256), 256, 0, stream>>>(t1, t2, tot, EXPD, LD1);
        }
        for (int ex = 0; ex < 2; ++ex) {
            const int* ridx = lists + ex * NTOK;
            const int* rc   = cnt + ex;
            const float* edw = rec_ed + (long)ex * EXPD * DMODEL;
            gemm(t1, LD1, edw, DMODEL, x_, DMODEL, NTOK, DMODEL, EXPD, 1, ridx, rc, stream);
        }

        // LTI state update: h = A*h + B*e + g_t*x
        lti_k<<<ewgrid, 256, 0, stream>>>(h_, e_, x_, lti_logA, lti_logdt,
                                          lti_B, gate_emb, t);
    }

    // ---- coda block ----
    run_attn(h_, h_, coda_an, coda_wq, coda_wk, coda_wv, coda_wo, coda_qn, coda_kn,
             t0, t1, kb, vb, t3, stream);
    run_swiglu(h_, h_, coda_fn, coda_g, coda_u, coda_d, t0, t1, t2, stream);

    // ---- final norm + lm_head ----
    rmsnorm_k<<<NTOK, 256, 0, stream>>>(h_, final_norm, t0, DMODEL);
    gemm(t0, DMODEL, lm_head, NVOCAB, (float*)d_out, NVOCAB,
         NTOK, NVOCAB, DMODEL, 0, nullptr, nullptr, stream);
}
